// DGCNN_54606214201506
// MI455X (gfx1250) — matrix operand order repaired and checked by reading, not compiled, not run on hardware
//
#include <hip/hip_runtime.h>
#include <hip/hip_bf16.h>

typedef __attribute__((ext_vector_type(16))) __bf16 v16bf;
typedef __attribute__((ext_vector_type(8)))  __bf16 v8bf;
typedef __attribute__((ext_vector_type(8)))  float  v8f;
typedef __attribute__((ext_vector_type(2)))  float  v2f;

#define KNNK 20
#define ECWAVES 2

__device__ __forceinline__ float lrelu(float h) { return h >= 0.f ? h : 0.2f * h; }

__device__ __forceinline__ void topk_insert(float (&tv)[KNNK], int (&ti)[KNNK], float d, int m) {
  if (d > tv[KNNK - 1]) {
    tv[KNNK - 1] = d; ti[KNNK - 1] = m;
#pragma unroll
    for (int t = KNNK - 1; t > 0; --t) {
      if (tv[t] > tv[t - 1]) {
        float fv = tv[t]; tv[t] = tv[t - 1]; tv[t - 1] = fv;
        int   fi = ti[t]; ti[t] = ti[t - 1]; ti[t - 1] = fi;
      }
    }
  }
}

// ---------------- KNN via fp32 WMMA (V_WMMA_F32_16X16X4_F32) ----------------
// Block = 64 threads (2 waves) owning 64 query rows; each wave owns 32 rows.
// Per 16-wide m-tile: Gram tiles by WMMA, distance via norms, LDS transpose,
// per-lane register top-20 insertion.
template <int C>
__global__ void __launch_bounds__(64)
knn_wmma_kernel(const float* __restrict__ x, int ldx, int N, int* __restrict__ idx_out)
{
  constexpr int CP = (C + 3) & ~3;   // K padded to multiple of 4 (zeros)
  constexpr int XP = CP + 2;         // LDS row pitch (8B aligned, bank-skewed)
  __shared__ float Xn[64][XP];
  __shared__ float xxn[64];
  __shared__ float Xm[16][XP];
  __shared__ float xq[16][4];
  __shared__ float xxm[16];
  __shared__ float dt[2][32][17];

  const int b   = blockIdx.y;
  const int n0  = blockIdx.x * 64;
  const int tid = threadIdx.x;
  const int lane = tid & 31, wv = tid >> 5;
  const int col = lane & 15, hi = lane >> 4;
  const int rowbase = wv * 32;

  // Stage the block's 64 query rows + norms
  {
    const float* __restrict__ xp = x + ((size_t)b * N + n0 + tid) * ldx;
    float s = 0.f;
#pragma unroll
    for (int c = 0; c < CP; ++c) {
      float vv = (c < C) ? xp[c] : 0.f;
      Xn[tid][c] = vv; s += vv * vv;
    }
    xxn[tid] = s;
  }
  __syncthreads();

  float xn0[8], xn1[8];
#pragma unroll
  for (int r = 0; r < 8; ++r) {
    xn0[r] = xxn[rowbase + 8 * hi + r];
    xn1[r] = xxn[rowbase + 16 + 8 * hi + r];
  }

  float tv[KNNK]; int ti[KNNK];
#pragma unroll
  for (int t = 0; t < KNNK; ++t) { tv[t] = -__builtin_inff(); ti[t] = 0; }

  for (int m0 = 0; m0 < N; m0 += 16) {
    __syncthreads();
    { // stage 16-row m-tile: thread (r,q) loads channels q, q+4, ...
      const int r = tid & 15, q = tid >> 4;
      const float* __restrict__ mp = x + ((size_t)b * N + m0 + r) * ldx;
      float s = 0.f;
      for (int c = q; c < CP; c += 4) {
        float vv = (c < C) ? mp[c] : 0.f;
        Xm[r][c] = vv; s += vv * vv;
      }
      xq[r][q] = s;
    }
    __syncthreads();
    if (tid < 16) xxm[tid] = xq[tid][0] + xq[tid][1] + xq[tid][2] + xq[tid][3];
    __syncthreads();

    // Gram tiles: rows [rowbase..+15] and [rowbase+16..+31] vs cols [m0..m0+15]
    v8f g0 = {}; v8f g1 = {};
#pragma unroll
    for (int k0 = 0; k0 < CP; k0 += 4) {
      // fp32 A 16x4: lane holds row=col, K = 2*hi + e ; B 4x16 mirrored
      v2f a0 = *(const v2f*)&Xn[rowbase + col][k0 + 2 * hi];
      v2f a1 = *(const v2f*)&Xn[rowbase + 16 + col][k0 + 2 * hi];
      v2f bm = *(const v2f*)&Xm[col][k0 + 2 * hi];
      g0 = __builtin_amdgcn_wmma_f32_16x16x4_f32(false, a0, false, bm, (short)0, g0, false, false);
      g1 = __builtin_amdgcn_wmma_f32_16x16x4_f32(false, a1, false, bm, (short)0, g1, false, false);
    }

    // dist = 2G - |xn|^2 - |xm|^2 ; transpose through LDS so each lane owns a row
    const float xmc = xxm[col];
#pragma unroll
    for (int r = 0; r < 8; ++r) {
      dt[wv][8 * hi + r][col]      = 2.f * g0[r] - xn0[r] - xmc;
      dt[wv][16 + 8 * hi + r][col] = 2.f * g1[r] - xn1[r] - xmc;
    }
#pragma unroll
    for (int j = 0; j < 16; ++j) topk_insert(tv, ti, dt[wv][lane][j], m0 + j);
  }

  int* __restrict__ op = idx_out + ((size_t)b * N + n0 + rowbase + lane) * KNNK;
#pragma unroll
  for (int t = 0; t < KNNK; ++t) op[t] = ti[t];
}

// ---------------- prep: bf16 weights (zero-padded K) + folded BN scale/shift ----------------
__global__ void prep_kernel(const float* __restrict__ W, const float* __restrict__ g,
                            const float* __restrict__ be, const float* __restrict__ mu,
                            const float* __restrict__ va,
                            __bf16* __restrict__ Wbf, float* __restrict__ sc, float* __restrict__ sh,
                            int cout, int cin, int K2)
{
  int t = blockIdx.x * blockDim.x + threadIdx.x;
  if (t < cout) {
    float s = g[t] * rsqrtf(va[t] + 1e-5f);
    sc[t] = s;
    sh[t] = be[t] - mu[t] * s;
  }
  if (t < cout * K2) {
    int o = t / K2, c = t % K2;
    Wbf[t] = (c < cin) ? (__bf16)W[o * cin + c] : (__bf16)0.0f;
  }
}

// ---------------- edge conv: one wave per point; 32 padded neighbor rows; bf16 WMMA ----------------
__global__ void __launch_bounds__(ECWAVES * 32)
edgeconv_kernel(const float* __restrict__ xin, int ldin, int C,
                const int* __restrict__ knn,
                const __bf16* __restrict__ Wbf, int K2,
                const float* __restrict__ bnsc, const float* __restrict__ bnsh,
                int cout,
                float* __restrict__ yout, __bf16* __restrict__ youtbf, int ldout, int N)
{
  extern __shared__ __bf16 smem[];
  const int lane = threadIdx.x & 31;
  const int wv   = threadIdx.x >> 5;
  const int p = blockIdx.x * ECWAVES + wv;
  const int b = p / N, n = p % N;
  const int FP = K2 + 16;                       // padded LDS row pitch (keeps 32B alignment)
  __bf16* feat = smem + (size_t)wv * 32 * FP;

  const float* __restrict__ xc = xin + ((size_t)b * N + n) * ldin;
  const int*   __restrict__ kp = knn + (size_t)p * KNNK;

  // Build 32 x K2 bf16 feature tile: rows 0..19 real neighbors, 20..31 duplicate row 0
  for (int j = 0; j < 32; ++j) {
    const int jj = (j < KNNK) ? j : 0;
    const int m = kp[jj];
    const float* __restrict__ xm = xin + ((size_t)b * N + m) * ldin;
    for (int c = lane; c < C; c += 32) {
      const float cen = xc[c];
      feat[j * FP + c]     = (__bf16)(xm[c] - cen);
      feat[j * FP + C + c] = (__bf16)cen;
    }
    for (int c = 2 * C + lane; c < K2; c += 32) feat[j * FP + c] = (__bf16)0.0f;
  }
  __syncthreads();

  const int col = lane & 15;
  const int hi  = lane >> 4;
  const int ntiles = cout >> 4;
  for (int nt = 0; nt < ntiles; ++nt) {
    const int o = nt * 16 + col;
    const float sc = bnsc[o], sh = bnsh[o];
    v8f acc0 = {}; v8f acc1 = {};
    for (int kk = 0; kk < K2; kk += 32) {
      // A operand (16-bit 16x32): lane holds row=col; K chunks at 8*hi and 16+8*hi
      v8bf a0l = *(const v8bf*)&feat[col * FP + kk + 8 * hi];
      v8bf a0h = *(const v8bf*)&feat[col * FP + kk + 16 + 8 * hi];
      v8bf a1l = *(const v8bf*)&feat[(16 + col) * FP + kk + 8 * hi];
      v8bf a1h = *(const v8bf*)&feat[(16 + col) * FP + kk + 16 + 8 * hi];
      v16bf a0 = __builtin_shufflevector(a0l, a0h, 0,1,2,3,4,5,6,7,8,9,10,11,12,13,14,15);
      v16bf a1 = __builtin_shufflevector(a1l, a1h, 0,1,2,3,4,5,6,7,8,9,10,11,12,13,14,15);
      // B operand (32x16): lane holds column o; K chunks at 8*hi and 16+8*hi
      v8bf bwl = *(const v8bf*)&Wbf[(size_t)o * K2 + kk + 8 * hi];
      v8bf bwh = *(const v8bf*)&Wbf[(size_t)o * K2 + kk + 16 + 8 * hi];
      v16bf bw = __builtin_shufflevector(bwl, bwh, 0,1,2,3,4,5,6,7,8,9,10,11,12,13,14,15);
      acc0 = __builtin_amdgcn_wmma_f32_16x16x32_bf16(false, a0, false, bw, (short)0, acc0, false, false);
      acc1 = __builtin_amdgcn_wmma_f32_16x16x32_bf16(false, a1, false, bw, (short)0, acc1, false, false);
    }
    // BN + LeakyReLU + max over the 32 rows (M dim): in-register then lane-xor-16
    float vmax = -__builtin_inff();
#pragma unroll
    for (int r = 0; r < 8; ++r) {
      vmax = fmaxf(vmax, fmaxf(lrelu(acc0[r] * sc + sh), lrelu(acc1[r] * sc + sh)));
    }
    vmax = fmaxf(vmax, __shfl_xor(vmax, 16, 32));
    if (lane < 16) {
      const size_t oidx = ((size_t)b * N + n) * ldout + o;
      yout[oidx]   = vmax;           // fp32 copy: consumed by next KNN / conv
      youtbf[oidx] = (__bf16)vmax;   // bf16 copy: consumed by the final WMMA GEMM
    }
  }
}

// ---------------- c5 GEMM + BN/LeakyReLU + global max & mean ----------------
__global__ void __launch_bounds__(128)
dgcnn_final(const __bf16* __restrict__ catbf,
            const __bf16* __restrict__ W5,
            const float* __restrict__ sc5, const float* __restrict__ sh5,
            float* __restrict__ out, int N)
{
  const int lane = threadIdx.x & 31;
  const int wv = threadIdx.x >> 5;
  const int gw = blockIdx.x * 4 + wv;   // B*64 waves total
  const int b  = gw >> 6;
  const int ot = gw & 63;
  const int col = lane & 15;
  const int hi  = lane >> 4;

  // Hoist A (W5 tile, M=o, K=512) into registers once
  v16bf afrag[16];
  const __bf16* __restrict__ wrow = W5 + (size_t)(ot * 16 + col) * 512;
#pragma unroll
  for (int kk = 0; kk < 16; ++kk) {
    v8bf wl = *(const v8bf*)(wrow + kk * 32 + 8 * hi);
    v8bf wh = *(const v8bf*)(wrow + kk * 32 + 16 + 8 * hi);
    afrag[kk] = __builtin_shufflevector(wl, wh, 0,1,2,3,4,5,6,7,8,9,10,11,12,13,14,15);
  }

  const int obase = ot * 16 + (hi << 3);
  float scv[8], shv[8], rmax[8], rsum[8];
#pragma unroll
  for (int r = 0; r < 8; ++r) {
    scv[r] = sc5[obase + r]; shv[r] = sh5[obase + r];
    rmax[r] = -__builtin_inff(); rsum[r] = 0.f;
  }

  for (int n0 = 0; n0 < N; n0 += 16) {
    v8f acc = {};
    const __bf16* __restrict__ crow = catbf + ((size_t)b * N + n0 + col) * 512 + 8 * hi;
#pragma unroll 4
    for (int kk = 0; kk < 16; ++kk) {
      v8bf bl = *(const v8bf*)(crow + kk * 32);
      v8bf bh = *(const v8bf*)(crow + kk * 32 + 16);
      v16bf bfrag = __builtin_shufflevector(bl, bh, 0,1,2,3,4,5,6,7,8,9,10,11,12,13,14,15);
      acc = __builtin_amdgcn_wmma_f32_16x16x32_bf16(false, afrag[kk], false, bfrag, (short)0, acc, false, false);
    }
#pragma unroll
    for (int r = 0; r < 8; ++r) {
      float h = lrelu(acc[r] * scv[r] + shv[r]);
      rmax[r] = fmaxf(rmax[r], h);
      rsum[r] += h;
    }
  }
#pragma unroll
  for (int r = 0; r < 8; ++r) {
    float mx = rmax[r], sm = rsum[r];
    for (int d = 1; d < 16; d <<= 1) {
      mx = fmaxf(mx, __shfl_xor(mx, d, 32));
      sm += __shfl_xor(sm, d, 32);
    }
    if (col == 0) {
      out[(size_t)b * 2048 + obase + r]        = mx;
      out[(size_t)b * 2048 + 1024 + obase + r] = sm * (1.0f / 2048.0f);
    }
  }
}

extern "C" void kernel_launch(void* const* d_in, const int* in_sizes, int n_in,
                              void* d_out, int out_size, void* d_ws, size_t ws_size,
                              hipStream_t stream)
{
  (void)n_in; (void)out_size; (void)ws_size;
  const int B = 16, N = 2048;

  // Robust to flattening order: x is the unique input with B*N*3 elements.
  int xidx = 0, pbase = 1;
  if (in_sizes[0] != B * N * 3) { xidx = 25; pbase = 0; }
  const float* x = (const float*)d_in[xidx];
  const float *W[5], *G[5], *Bt[5], *Mu[5], *Va[5];
  for (int l = 0; l < 5; ++l) {
    W[l]  = (const float*)d_in[pbase + 5 * l + 0];
    G[l]  = (const float*)d_in[pbase + 5 * l + 1];
    Bt[l] = (const float*)d_in[pbase + 5 * l + 2];
    Mu[l] = (const float*)d_in[pbase + 5 * l + 3];
    Va[l] = (const float*)d_in[pbase + 5 * l + 4];
  }
  static const int cout[5] = {64, 64, 128, 256, 1024};
  static const int cin2[5] = {6, 128, 128, 256, 512};   // real K (=2C, or 512 for c5)
  static const int K2[5]   = {32, 128, 128, 256, 512};  // padded K (multiple of 32)

  char* ws = (char*)d_ws;
  size_t off = 0;
  auto walloc = [&](size_t bytes) -> void* {
    void* pp = ws + off;
    off += (bytes + 255) & ~(size_t)255;
    return pp;
  };
  float*  cat   = (float*) walloc((size_t)B * N * 512 * sizeof(float));   // fp32: KNN input
  __bf16* catbf = (__bf16*)walloc((size_t)B * N * 512 * sizeof(__bf16));  // bf16: final GEMM input
  int*  knnidx  = (int*)   walloc((size_t)B * N * KNNK * sizeof(int));
  __bf16* Wbf[5]; float* SC[5]; float* SH[5];
  for (int l = 0; l < 5; ++l) {
    Wbf[l] = (__bf16*)walloc((size_t)cout[l] * K2[l] * sizeof(__bf16));
    SC[l]  = (float*) walloc((size_t)cout[l] * sizeof(float));
    SH[l]  = (float*) walloc((size_t)cout[l] * sizeof(float));
  }

  for (int l = 0; l < 5; ++l) {
    int tot = cout[l] * K2[l];
    prep_kernel<<<(tot + 255) / 256, 256, 0, stream>>>(W[l], G[l], Bt[l], Mu[l], Va[l],
                                                       Wbf[l], SC[l], SH[l], cout[l], cin2[l], K2[l]);
  }

  dim3 kgrid(N / 64, B);
  const int ecgrid = (B * N) / ECWAVES;
  auto ecsmem = [](int k2) { return (size_t)ECWAVES * 32 * (k2 + 16) * sizeof(__bf16); };

  // Layer 1: input x (B,N,3)
  knn_wmma_kernel<3><<<kgrid, 64, 0, stream>>>(x, 3, N, knnidx);
  edgeconv_kernel<<<ecgrid, ECWAVES * 32, ecsmem(K2[0]), stream>>>(
      x, 3, 3, knnidx, Wbf[0], K2[0], SC[0], SH[0], cout[0], cat + 0, catbf + 0, 512, N);
  // Layer 2: input cat[:, :, 0:64]
  knn_wmma_kernel<64><<<kgrid, 64, 0, stream>>>(cat + 0, 512, N, knnidx);
  edgeconv_kernel<<<ecgrid, ECWAVES * 32, ecsmem(K2[1]), stream>>>(
      cat + 0, 512, 64, knnidx, Wbf[1], K2[1], SC[1], SH[1], cout[1], cat + 64, catbf + 64, 512, N);
  // Layer 3: input cat[:, :, 64:128]
  knn_wmma_kernel<64><<<kgrid, 64, 0, stream>>>(cat + 64, 512, N, knnidx);
  edgeconv_kernel<<<ecgrid, ECWAVES * 32, ecsmem(K2[2]), stream>>>(
      cat + 64, 512, 64, knnidx, Wbf[2], K2[2], SC[2], SH[2], cout[2], cat + 128, catbf + 128, 512, N);
  // Layer 4: input cat[:, :, 128:256]
  knn_wmma_kernel<128><<<kgrid, 64, 0, stream>>>(cat + 128, 512, N, knnidx);
  edgeconv_kernel<<<ecgrid, ECWAVES * 32, ecsmem(K2[3]), stream>>>(
      cat + 128, 512, 128, knnidx, Wbf[3], K2[3], SC[3], SH[3], cout[3], cat + 256, catbf + 256, 512, N);

  // Final: c5 GEMM + BN/LReLU + global max/mean -> (B, 2048)
  dgcnn_final<<<(B * 64) / 4, 128, 0, stream>>>(catbf, Wbf[4], SC[4], SH[4], (float*)d_out, N);
}
